// PAM_Module_83691732730337
// MI455X (gfx1250) — 16-bit and 8-bit matrix operand order repaired and checked by reading, compiled, not run on hardware
//
#include <hip/hip_runtime.h>
#include <hip/hip_bf16.h>
#include <math.h>

typedef __bf16 bf16;
typedef __attribute__((ext_vector_type(8)))  bf16  v8bf;
typedef __attribute__((ext_vector_type(16))) bf16  v16bf;
typedef __attribute__((ext_vector_type(8)))  float v8f;
typedef __attribute__((ext_vector_type(8)))  int   v8i;

union V16U { v16bf v; v8bf h[2]; };
union V8I  { v8i v; unsigned long long q[4]; int4 x[2]; };

static __device__ inline v8f wmma_bf16(v16bf a, v16bf b, v8f c) {
    return __builtin_amdgcn_wmma_f32_16x16x32_bf16(false, a, false, b, (short)0, c, false, false);
}
static __device__ inline v8f wmma_fp8(v8i a, v8i b, v8f c) {
    return __builtin_amdgcn_wmma_f32_16x16x64_fp8_fp8(a, b, (short)0, c, false, false);
}

static __device__ inline unsigned int sw_fp8(float f) {
    unsigned int s = (__float_as_uint(f) >> 24) & 0x80u;
    float a = fminf(fabsf(f), 448.f);
    unsigned int x = __float_as_uint(a);
    unsigned int rx = x + 0x7FFFFu + ((x >> 20) & 1u);
    int e = (int)(rx >> 23) - 127;
    if (e < -6) return s | (unsigned int)rintf(a * 512.f);
    return s | ((unsigned int)(e + 7) << 3) | ((rx >> 20) & 7u);
}
static __device__ inline unsigned int pack_fp8x4(float a, float b, float c, float d) {
#if __has_builtin(__builtin_amdgcn_cvt_pk_fp8_f32)
    int w = __builtin_amdgcn_cvt_pk_fp8_f32(a, b, 0, false);
    w = __builtin_amdgcn_cvt_pk_fp8_f32(c, d, w, true);
    return (unsigned int)w;
#else
    return sw_fp8(a) | (sw_fp8(b) << 8) | (sw_fp8(c) << 16) | (sw_fp8(d) << 24);
#endif
}
static __device__ inline unsigned char fp8_1(float a) {
    return (unsigned char)(pack_fp8x4(a, a, a, a) & 0xffu);
}

#define CHW  512
#define SSTR 512
#define PSTR 520

__global__ __launch_bounds__(256) void pam_proj_kernel(
    const float* __restrict__ x2d,
    const float* __restrict__ Wq, const float* __restrict__ bq,
    const float* __restrict__ Wk, const float* __restrict__ bk,
    const float* __restrict__ Wv, const float* __restrict__ bv,
    bf16* __restrict__ Qb, bf16* __restrict__ Kb, unsigned char* __restrict__ Vb8)
{
    const int j = blockIdx.x * 256 + threadIdx.x;
    const int b = blockIdx.y;
    const int n = j >> 4;
    const int d = j & 15;
    const float* X = x2d + (size_t)b * 64 * 65536;

    float aq[8] = {}, ak[8] = {}, av[64] = {};
    for (int c = 0; c < 64; ++c) {
        const float xv = X[(size_t)c * 65536 + j];
        #pragma unroll
        for (int o = 0; o < 8; ++o)  aq[o] = fmaf(Wq[o * 64 + c], xv, aq[o]);
        #pragma unroll
        for (int o = 0; o < 8; ++o)  ak[o] = fmaf(Wk[o * 64 + c], xv, ak[o]);
        #pragma unroll
        for (int o = 0; o < 64; ++o) av[o] = fmaf(Wv[o * 64 + c], xv, av[o]);
    }

    bf16* q = Qb + ((size_t)b * 4096 + n) * 128 + d;
    bf16* k = Kb + ((size_t)b * 4096 + n) * 128 + d;
    #pragma unroll
    for (int o = 0; o < 8; ++o) {
        q[o * 16] = (bf16)(aq[o] + bq[o]);
        k[o * 16] = (bf16)(ak[o] + bk[o]);
    }
    #pragma unroll
    for (int o = 0; o < 64; ++o)
        Vb8[((size_t)b * 1024 + o * 16 + d) * 4096 + n] = fp8_1(av[o] + bv[o]);
}

__global__ __launch_bounds__(256) void pam_attn_kernel(
    const bf16* __restrict__ Qb, const bf16* __restrict__ Kb,
    const unsigned char* __restrict__ Vb8,
    const float* __restrict__ x3d, const float* __restrict__ gamma_p,
    float* __restrict__ out)
{
    __shared__ __align__(32) bf16 Sc[16 * SSTR];
    __shared__ __align__(16) unsigned char Pc8[2][16 * PSTR];
    __shared__ float red[16 * 16];
    __shared__ float rowmax[2][16], rowsum[2][16], rowscale[2][16];

    const int tid  = threadIdx.x;
    const int wave = tid >> 5;
    const int lane = tid & 31;
    const int half = lane >> 4;
    const int lrow = lane & 15;
    const int nblk = blockIdx.x;
    const int b    = blockIdx.y;

    const bf16* Q = Qb + (size_t)b * 4096 * 128;
    const bf16* K = Kb + (size_t)b * 4096 * 128;
    const unsigned char* V = Vb8 + (size_t)b * 1024 * 4096;

    if (tid < 32) { rowmax[tid >> 4][tid & 15] = -INFINITY; rowsum[tid >> 4][tid & 15] = 0.f; }

    v8f acc[2][8] = {};
    __syncthreads();

    for (int ch = 0; ch < 8; ++ch) {
        const int mbase = ch * CHW;
        const int mnext = ((ch + 1) & 7) * CHW;

        for (int nb = 0; nb < 2; ++nb) {
            V16U aq[4];
            #pragma unroll
            for (int ks = 0; ks < 4; ++ks) {
                const bf16* qp = Q + ((size_t)(nblk * 32 + nb * 16 + lrow)) * 128
                                   + ks * 32 + half * 16;
                aq[ks].h[0] = *(const v8bf*)(qp);
                aq[ks].h[1] = *(const v8bf*)(qp + 8);
            }
            #pragma unroll 1
            for (int t = 0; t < 4; ++t) {
                const int m0 = wave * 64 + t * 16;
                const bf16* kp = K + ((size_t)(mbase + m0 + lrow)) * 128 + half * 16;
                if (nb == 0)
                    __builtin_prefetch(K + ((size_t)(mnext + m0 + lrow)) * 128, 0, 3);
                v16bf k0 = *(const v16bf*)(kp +  0);
                v16bf k1 = *(const v16bf*)(kp + 32);
                v16bf k2 = *(const v16bf*)(kp + 64);
                v16bf k3 = *(const v16bf*)(kp + 96);
                v8f s = {};
                s = wmma_bf16(aq[0].v, k0, s);
                s = wmma_bf16(aq[1].v, k1, s);
                s = wmma_bf16(aq[2].v, k2, s);
                s = wmma_bf16(aq[3].v, k3, s);
                #pragma unroll
                for (int r = 0; r < 8; ++r)
                    Sc[(r + 8 * half) * SSTR + m0 + lrow] = (bf16)s[r];
            }
            __syncthreads();

            {
                const int r = tid >> 4, jj = tid & 15;
                const bf16* srow = &Sc[r * SSTR];
                float mx = -INFINITY;
                #pragma unroll 1
                for (int i = 0; i < 4; ++i) {
                    const v8bf vv = *(const v8bf*)(srow + jj * 8 + i * 128);
                    #pragma unroll
                    for (int e = 0; e < 8; ++e) mx = fmaxf(mx, (float)vv[e]);
                }
                red[r * 16 + jj] = mx;
                __syncthreads();
                if (tid < 16) {
                    float cm = red[tid * 16];
                    #pragma unroll
                    for (int i = 1; i < 16; ++i) cm = fmaxf(cm, red[tid * 16 + i]);
                    const float om = rowmax[nb][tid];
                    const float nm = fmaxf(om, cm);
                    rowscale[nb][tid] = __expf(om - nm);
                    rowmax[nb][tid] = nm;
                }
                __syncthreads();
                const float nm = rowmax[nb][r];
                float sm = 0.f;
                unsigned char* prow = &Pc8[nb][r * PSTR];
                #pragma unroll 1
                for (int i = 0; i < 4; ++i) {
                    const int col = jj * 8 + i * 128;
                    const v8bf vv = *(const v8bf*)(srow + col);
                    float p[8];
                    #pragma unroll
                    for (int e = 0; e < 8; ++e) {
                        p[e] = __expf((float)vv[e] - nm);
                        sm += p[e];
                    }
                    uint2 w;
                    w.x = pack_fp8x4(p[0], p[1], p[2], p[3]);
                    w.y = pack_fp8x4(p[4], p[5], p[6], p[7]);
                    *(uint2*)(prow + col) = w;
                }
                red[r * 16 + jj] = sm;
                __syncthreads();
                if (tid < 16) {
                    float cs = 0.f;
                    #pragma unroll
                    for (int i = 0; i < 16; ++i) cs += red[tid * 16 + i];
                    rowsum[nb][tid] = rowsum[nb][tid] * rowscale[nb][tid] + cs;
                }
                __syncthreads();
            }
        }

        {
            #pragma unroll
            for (int nb = 0; nb < 2; ++nb) {
                #pragma unroll
                for (int r = 0; r < 8; ++r) {
                    const float scl = rowscale[nb][r + 8 * half];
                    #pragma unroll
                    for (int t = 0; t < 8; ++t) acc[nb][t][r] *= scl;
                }
            }
            const unsigned char* vb =
                V + ((size_t)(wave * 128 + lrow)) * 4096 + mbase + half * 8;
            const unsigned char* pr0 = &Pc8[0][lrow * PSTR + half * 8];
            const unsigned char* pr1 = &Pc8[1][lrow * PSTR + half * 8];
            #pragma unroll 1
            for (int ms = 0; ms < 8; ++ms) {
                V8I a0, a1;
                const unsigned char* p0 = pr0 + ms * 64;
                const unsigned char* p1 = pr1 + ms * 64;
                a0.q[0] = *(const unsigned long long*)(p0);
                a0.q[1] = *(const unsigned long long*)(p0 + 16);
                a0.q[2] = *(const unsigned long long*)(p0 + 32);
                a0.q[3] = *(const unsigned long long*)(p0 + 48);
                a1.q[0] = *(const unsigned long long*)(p1);
                a1.q[1] = *(const unsigned long long*)(p1 + 16);
                a1.q[2] = *(const unsigned long long*)(p1 + 32);
                a1.q[3] = *(const unsigned long long*)(p1 + 48);
                V8I bf[8];
                #pragma unroll
                for (int t = 0; t < 8; ++t) {
                    const unsigned char* vp = vb + (size_t)t * 16 * 4096 + ms * 64;
                    bf[t].q[0] = *(const unsigned long long*)(vp);
                    bf[t].q[1] = *(const unsigned long long*)(vp + 16);
                    bf[t].q[2] = *(const unsigned long long*)(vp + 32);
                    bf[t].q[3] = *(const unsigned long long*)(vp + 48);
                }
                #pragma unroll
                for (int t = 0; t < 8; ++t) {
                    acc[0][t] = wmma_fp8(a0.v, bf[t].v, acc[0][t]);
                    acc[1][t] = wmma_fp8(a1.v, bf[t].v, acc[1][t]);
                }
            }
            __syncthreads();
        }
    }

    const float g = gamma_p[0];
    const float* xr = x3d + (size_t)b * 64 * 65536;
    float* op = out + (size_t)b * 64 * 65536;
    #pragma unroll 1
    for (int t = 0; t < 8; ++t) {
        const int cp = wave * 128 + t * 16 + lrow;
        const int c  = cp >> 4;
        const int d  = cp & 15;
        #pragma unroll
        for (int nb = 0; nb < 2; ++nb) {
            #pragma unroll
            for (int r = 0; r < 8; ++r) {
                const int nl = r + 8 * half;
                const int n  = nblk * 32 + nb * 16 + nl;
                const float inv = 1.f / rowsum[nb][nl];
                const size_t o = (size_t)c * 65536 + (size_t)n * 16 + d;
                op[o] = g * acc[nb][t][r] * inv + xr[o];
            }
        }
    }
}

extern "C" void kernel_launch(void* const* d_in, const int* in_sizes, int n_in,
                              void* d_out, int out_size, void* d_ws, size_t ws_size,
                              hipStream_t stream) {
    const float* x2d = (const float*)d_in[0];
    const float* x3d = (const float*)d_in[1];
    const float* Wq  = (const float*)d_in[2];
    const float* bq  = (const float*)d_in[3];
    const float* Wk  = (const float*)d_in[4];
    const float* bk  = (const float*)d_in[5];
    const float* Wv  = (const float*)d_in[6];
    const float* bv  = (const float*)d_in[7];
    const float* gm  = (const float*)d_in[8];
    float* out = (float*)d_out;

    bf16* Qb = (bf16*)d_ws;
    bf16* Kb = (bf16*)((char*)d_ws + (size_t)(2u << 20));
    unsigned char* Vb8 = (unsigned char*)d_ws + (size_t)(4u << 20);

    pam_proj_kernel<<<dim3(256, 2), 256, 0, stream>>>(
        x2d, Wq, bq, Wk, bk, Wv, bv, Qb, Kb, Vb8);
    pam_attn_kernel<<<dim3(128, 2), 256, 0, stream>>>(
        Qb, Kb, Vb8, x3d, gm, out);
}
